// GraphSageWithSampling_36816459662034
// MI455X (gfx1250) — matrix operand order repaired and checked by reading, not compiled, not run on hardware
//
#include <hip/hip_runtime.h>
#include <hip/hip_bf16.h>
#include <cstdint>

typedef __attribute__((ext_vector_type(16))) __bf16 v16bf;
typedef __attribute__((ext_vector_type(2)))  __bf16 v2bf;
typedef __attribute__((ext_vector_type(8)))  float  v8f;

#define FDIM 128
#define CDIM 300
#define CPAD 320

// f32 -> bf16 pair, packed into one u32. Plain fptrunc (RTNE) lets the
// backend pick the native cvt instruction if gfx1250 has one.
__device__ __forceinline__ unsigned pack2bf(float lo, float hi) {
  union { v2bf v; unsigned u; } u;
  u.v[0] = (__bf16)lo;
  u.v[1] = (__bf16)hi;
  return u.u;
}

// Load one 16x32(bf16) A or 32x16(bf16) B fragment as two 16B LDS chunks.
__device__ __forceinline__ v16bf ldfrag(const unsigned short* p0,
                                        const unsigned short* p1) {
  union { v16bf v; uint4 q[2]; } u;
  u.q[0] = *reinterpret_cast<const uint4*>(p0);
  u.q[1] = *reinterpret_cast<const uint4*>(p1);
  return u.v;
}

__device__ __forceinline__ float leaky(float x) { return x > 0.f ? x : 0.01f * x; }

// ---------------------------------------------------------------------------
// One-time weight conversion: f32 [rows][cols] -> packed bf16 [rows][padcols],
// zero-padded past cols. Tiny kernel; guards here are irrelevant.
// ---------------------------------------------------------------------------
__global__ __launch_bounds__(256) void wcvt_kernel(
    const float* __restrict__ W, unsigned* __restrict__ out,
    int rows, int cols, int padcols)
{
  int t = blockIdx.x * 256 + threadIdx.x;
  int ppr = padcols >> 1;
  if (t >= rows * ppr) return;
  int row = t / ppr, cp = t - row * ppr;
  int k = cp * 2;
  float lo = (k < cols) ? W[(size_t)row * cols + k] : 0.f;
  float hi = (k + 1 < cols) ? W[(size_t)row * cols + k + 1] : 0.f;
  out[t] = pack2bf(lo, hi);
}

// ---------------------------------------------------------------------------
// h = node_emb[nid+1] + leaky_relu(content @ Wp^T + bp)
// 64-row x 128-col tile per block; 8 waves, one 16-wide N-tile per wave.
// K = 300 (10 x 32 steps; B pre-padded to 320; A tail handled by select).
// LDS tiles packed u32 (2 bf16), row stride 20 u32 (= 40 halfs).
// ---------------------------------------------------------------------------
__global__ __launch_bounds__(256) void embed_kernel(
    const float* __restrict__ content, const int* __restrict__ nid,
    const float* __restrict__ emb, const unsigned* __restrict__ Wpb,
    const float* __restrict__ bp, float* __restrict__ out)
{
  __shared__ __align__(16) unsigned As[64 * 20];
  __shared__ __align__(16) unsigned Bs[128 * 20];
  const unsigned short* Asu = reinterpret_cast<const unsigned short*>(As);
  const unsigned short* Bsu = reinterpret_cast<const unsigned short*>(Bs);
  const int tid  = threadIdx.x;
  const int lane = tid & 31, wv = tid >> 5;
  const int half = lane >> 4, nlan = lane & 15;
  const int row0 = blockIdx.x * 64;
  const bool lastblk = (blockIdx.x == gridDim.x - 1);

  v8f acc[4] = {};

  const int ar = tid >> 2, akj = (tid & 3) * 8;      // A stage: 64 rows x 32 k
  const int br = tid >> 1, bkj = (tid & 1) * 16;     // B stage: 128 rows x 32 k
  const float* gArow = content + (size_t)(row0 + ar) * CDIM;
  const unsigned* gBrow = Wpb + (size_t)br * (CPAD / 2) + (bkj >> 1);
  unsigned* Adst = &As[ar * 20 + (akj >> 1)];
  unsigned* Bdst = &Bs[br * 20 + (bkj >> 1)];

  for (int kt = 0; kt < 10; ++kt) {
    const int k0 = kt * 32;
    __syncthreads();
    // ---- B stage: pure copy of pre-converted, pre-padded bf16 weights ----
    {
      const uint4* b4 = reinterpret_cast<const uint4*>(gBrow + (k0 >> 1));
      uint4 pb0 = b4[0], pb1 = b4[1];
      *reinterpret_cast<uint4*>(Bdst)     = pb0;
      *reinterpret_cast<uint4*>(Bdst + 4) = pb1;
    }
    // ---- A stage ----
    if (kt < 9) {
      __builtin_prefetch(gArow + k0 + akj + 32, 0, 1);
      const float4* a4 = reinterpret_cast<const float4*>(gArow + k0 + akj);
      float4 a0 = a4[0], a1 = a4[1];
      uint4 pa;
      pa.x = pack2bf(a0.x, a0.y); pa.y = pack2bf(a0.z, a0.w);
      pa.z = pack2bf(a1.x, a1.y); pa.w = pack2bf(a1.z, a1.w);
      *reinterpret_cast<uint4*>(Adst) = pa;
    } else if (!lastblk) {
      // tail (k0=288): unconditional vector loads (over-read stays in-bounds,
      // lands in the next row), then zero invalid K lanes by value-select.
      const float4* a4 = reinterpret_cast<const float4*>(gArow + k0 + akj);
      float4 a0 = a4[0], a1 = a4[1];
      float va[8] = {a0.x, a0.y, a0.z, a0.w, a1.x, a1.y, a1.z, a1.w};
#pragma unroll
      for (int i = 0; i < 8; ++i) {
        int k = k0 + akj + i;
        va[i] = (k < CDIM) ? va[i] : 0.f;
      }
      uint4 pa;
      pa.x = pack2bf(va[0], va[1]); pa.y = pack2bf(va[2], va[3]);
      pa.z = pack2bf(va[4], va[5]); pa.w = pack2bf(va[6], va[7]);
      *reinterpret_cast<uint4*>(Adst) = pa;
    } else {
      // very last block: clamped scalar loads (never past buffer end)
      float va[8];
#pragma unroll
      for (int i = 0; i < 8; ++i) {
        int k  = k0 + akj + i;
        int kc = k < CDIM ? k : (CDIM - 1);
        float v = gArow[kc];
        va[i] = (k < CDIM) ? v : 0.f;
      }
      uint4 pa;
      pa.x = pack2bf(va[0], va[1]); pa.y = pack2bf(va[2], va[3]);
      pa.z = pack2bf(va[4], va[5]); pa.w = pack2bf(va[6], va[7]);
      *reinterpret_cast<uint4*>(Adst) = pa;
    }
    __syncthreads();
    // B fragment: lane n holds column (16*wv+n), K in {half*8..half*8+7, 16+half*8..16+half*8+7}
    const unsigned short* bptr = &Bsu[(16 * wv + nlan) * 40 + half * 8];
    v16bf bf = ldfrag(bptr, bptr + 16);
#pragma unroll
    for (int m = 0; m < 4; ++m) {
      // A fragment: lanes 0-15 K in {0..7,16..23}; lanes 16-31 K in {8..15,24..31}
      const unsigned short* arow = &Asu[(16 * m + nlan) * 40];
      v16bf af = ldfrag(arow + half * 8, arow + 16 + half * 8);
      acc[m] = __builtin_amdgcn_wmma_f32_16x16x32_bf16(
          false, af, false, bf, (short)0, acc[m], false, false);
    }
  }

  const int col = 16 * wv + nlan;
  const float bias = bp[col];
#pragma unroll
  for (int m = 0; m < 4; ++m) {
#pragma unroll
    for (int r = 0; r < 8; ++r) {
      int row = row0 + 16 * m + 8 * half + r;
      float v = leaky(acc[m][r] + bias);
      v += emb[(size_t)(nid[row] + 1) * FDIM + col];
      out[(size_t)row * FDIM + col] = v;
    }
  }
}

// ---------------------------------------------------------------------------
// Edge scatter: agg[dst] += h_src[src]; cnt[dst] += 1   (one thread per (e,f))
// ---------------------------------------------------------------------------
__global__ __launch_bounds__(256) void scatter_kernel(
    const float* __restrict__ hsrc, const int* __restrict__ src,
    const int* __restrict__ dst, float* __restrict__ agg,
    float* __restrict__ cnt, int E)
{
  long long t = (long long)blockIdx.x * 256 + threadIdx.x;
  if (t >= (long long)E * FDIM) return;
  int e = (int)(t >> 7);
  int f = (int)(t & 127);
  int s = src[e], d = dst[e];
  atomicAdd(&agg[(size_t)d * FDIM + f], hsrc[(size_t)s * FDIM + f]);
  if (f == 0) atomicAdd(&cnt[d], 1.0f);
}

// ---------------------------------------------------------------------------
// h_new = normalize(leaky_relu([h_self | (agg-h_self)/max(w-1,1)] @ Wc^T + bc))
// K = 256 exact (8 unguarded k-steps); same WMMA tiling as embed.
// ---------------------------------------------------------------------------
__global__ __launch_bounds__(256) void conv_kernel(
    const float* __restrict__ hself, const float* __restrict__ agg,
    const float* __restrict__ cnt, const unsigned* __restrict__ Wcb,
    const float* __restrict__ bc, float* __restrict__ out)
{
  __shared__ __align__(16) unsigned As[64 * 20];
  __shared__ __align__(16) unsigned Bs[128 * 20];
  __shared__ float rden[64];
  __shared__ float outl[64 * 132];
  const unsigned short* Asu = reinterpret_cast<const unsigned short*>(As);
  const unsigned short* Bsu = reinterpret_cast<const unsigned short*>(Bs);
  const int tid  = threadIdx.x;
  const int lane = tid & 31, wv = tid >> 5;
  const int half = lane >> 4, nlan = lane & 15;
  const int row0 = blockIdx.x * 64;

  if (tid < 64) {
    float c = cnt[row0 + tid];
    rden[tid] = 1.0f / fmaxf(c - 1.0f, 1.0f);
  }

  v8f acc[4] = {};
  const int ar = tid >> 2, akj = (tid & 3) * 8;
  const int br = tid >> 1, bkj = (tid & 1) * 16;
  const float* gS = hself + (size_t)(row0 + ar) * FDIM;
  const float* gG = agg + (size_t)(row0 + ar) * FDIM;
  const unsigned* gBrow = Wcb + (size_t)br * 128 + (bkj >> 1);
  unsigned* Adst = &As[ar * 20 + (akj >> 1)];
  unsigned* Bdst = &Bs[br * 20 + (bkj >> 1)];

  for (int kt = 0; kt < 8; ++kt) {
    const int k0 = kt * 32;
    __syncthreads();
    {   // B stage: copy pre-converted bf16 weights
      const uint4* b4 = reinterpret_cast<const uint4*>(gBrow + (k0 >> 1));
      uint4 pb0 = b4[0], pb1 = b4[1];
      *reinterpret_cast<uint4*>(Bdst)     = pb0;
      *reinterpret_cast<uint4*>(Bdst + 4) = pb1;
    }
    if (kt < 4) {                                // A cols 0..127 = h_self
      const float4* s4 = reinterpret_cast<const float4*>(gS + k0 + akj);
      float4 s0 = s4[0], s1 = s4[1];
      uint4 pa;
      pa.x = pack2bf(s0.x, s0.y); pa.y = pack2bf(s0.z, s0.w);
      pa.z = pack2bf(s1.x, s1.y); pa.w = pack2bf(s1.z, s1.w);
      *reinterpret_cast<uint4*>(Adst) = pa;
    } else {                                     // A cols 128..255 = (agg-hs)*rd
      float rd = rden[ar];
      const int kk = k0 - 128 + akj;
      const float4* s4 = reinterpret_cast<const float4*>(gS + kk);
      const float4* g4 = reinterpret_cast<const float4*>(gG + kk);
      float4 s0 = s4[0], s1 = s4[1], g0 = g4[0], g1 = g4[1];
      uint4 pa;
      pa.x = pack2bf((g0.x - s0.x) * rd, (g0.y - s0.y) * rd);
      pa.y = pack2bf((g0.z - s0.z) * rd, (g0.w - s0.w) * rd);
      pa.z = pack2bf((g1.x - s1.x) * rd, (g1.y - s1.y) * rd);
      pa.w = pack2bf((g1.z - s1.z) * rd, (g1.w - s1.w) * rd);
      *reinterpret_cast<uint4*>(Adst) = pa;
    }
    __syncthreads();
    const unsigned short* bptr = &Bsu[(16 * wv + nlan) * 40 + half * 8];
    v16bf bf = ldfrag(bptr, bptr + 16);
#pragma unroll
    for (int m = 0; m < 4; ++m) {
      const unsigned short* arow = &Asu[(16 * m + nlan) * 40];
      v16bf af = ldfrag(arow + half * 8, arow + 16 + half * 8);
      acc[m] = __builtin_amdgcn_wmma_f32_16x16x32_bf16(
          false, af, false, bf, (short)0, acc[m], false, false);
    }
  }

  // Epilogue: bias + leaky into LDS, then per-row L2 normalize.
  const int col = 16 * wv + nlan;
  const float bias = bc[col];
#pragma unroll
  for (int m = 0; m < 4; ++m) {
#pragma unroll
    for (int r = 0; r < 8; ++r) {
      int rt = 16 * m + 8 * half + r;
      outl[rt * 132 + col] = leaky(acc[m][r] + bias);
    }
  }
  __syncthreads();
  if (tid < 64) {
    float ss = 0.f;
#pragma unroll 4
    for (int c = 0; c < FDIM; ++c) { float v = outl[tid * 132 + c]; ss += v * v; }
    rden[tid] = 1.0f / fmaxf(sqrtf(ss), 1e-6f);
  }
  __syncthreads();
#pragma unroll
  for (int i = 0; i < 32; ++i) {
    int e = tid * 32 + i;
    int rt = e >> 7, c = e & 127;
    out[(size_t)(row0 + rt) * FDIM + c] = outl[rt * 132 + c] * rden[rt];
  }
}

// ---------------------------------------------------------------------------
extern "C" void kernel_launch(void* const* d_in, const int* in_sizes, int n_in,
                              void* d_out, int out_size, void* d_ws, size_t ws_size,
                              hipStream_t stream)
{
  const int*   nid0     = (const int*)d_in[0];
  const int*   nid1     = (const int*)d_in[1];
  const int*   nid2     = (const int*)d_in[2];
  const float* content0 = (const float*)d_in[3];
  const float* content1 = (const float*)d_in[4];
  const float* content2 = (const float*)d_in[5];
  const int*   b0s      = (const int*)d_in[6];
  const int*   b0d      = (const int*)d_in[7];
  const int*   b1s      = (const int*)d_in[8];
  const int*   b1d      = (const int*)d_in[9];
  const float* emb      = (const float*)d_in[10];
  const float* Wp0      = (const float*)d_in[11];
  const float* bp0      = (const float*)d_in[12];
  const float* Wp1      = (const float*)d_in[13];
  const float* bp1      = (const float*)d_in[14];
  const float* Wp2      = (const float*)d_in[15];
  const float* bp2      = (const float*)d_in[16];
  const float* Wc0      = (const float*)d_in[17];
  const float* bc0      = (const float*)d_in[18];
  const float* Wc1      = (const float*)d_in[19];
  const float* bc1      = (const float*)d_in[20];

  const int N0 = 409600, N1 = 40960, N2 = 4096, E0 = 409600, E1 = 40960;

  char* ws = (char*)d_ws;
  float* h0   = (float*)ws; ws += (size_t)N0 * FDIM * 4;
  float* h1e  = (float*)ws; ws += (size_t)N1 * FDIM * 4;
  float* h1   = (float*)ws; ws += (size_t)N1 * FDIM * 4;
  float* h2e  = (float*)ws; ws += (size_t)N2 * FDIM * 4;
  float* agg0 = (float*)ws; ws += (size_t)N1 * FDIM * 4;   // zeroed region starts here
  float* agg1 = (float*)ws; ws += (size_t)N2 * FDIM * 4;
  float* cnt0 = (float*)ws; ws += (size_t)N1 * 4;
  float* cnt1 = (float*)ws; ws += (size_t)N2 * 4;
  unsigned* wpb0 = (unsigned*)ws; ws += (size_t)FDIM * (CPAD / 2) * 4;
  unsigned* wpb1 = (unsigned*)ws; ws += (size_t)FDIM * (CPAD / 2) * 4;
  unsigned* wpb2 = (unsigned*)ws; ws += (size_t)FDIM * (CPAD / 2) * 4;
  unsigned* wcb0 = (unsigned*)ws; ws += (size_t)FDIM * 128 * 4;
  unsigned* wcb1 = (unsigned*)ws; ws += (size_t)FDIM * 128 * 4;

  size_t zbytes = (size_t)N1 * FDIM * 4 + (size_t)N2 * FDIM * 4 +
                  (size_t)N1 * 4 + (size_t)N2 * 4;
  hipMemsetAsync(agg0, 0, zbytes, stream);   // agg0|agg1|cnt0|cnt1 contiguous

  // One-time weight conversion to packed, zero-padded bf16.
  const int pW = FDIM * (CPAD / 2), pC = FDIM * 128;
  wcvt_kernel<<<(pW + 255) / 256, 256, 0, stream>>>(Wp0, wpb0, FDIM, CDIM, CPAD);
  wcvt_kernel<<<(pW + 255) / 256, 256, 0, stream>>>(Wp1, wpb1, FDIM, CDIM, CPAD);
  wcvt_kernel<<<(pW + 255) / 256, 256, 0, stream>>>(Wp2, wpb2, FDIM, CDIM, CPAD);
  wcvt_kernel<<<(pC + 255) / 256, 256, 0, stream>>>(Wc0, wcb0, FDIM, 256, 256);
  wcvt_kernel<<<(pC + 255) / 256, 256, 0, stream>>>(Wc1, wcb1, FDIM, 256, 256);

  embed_kernel<<<N0 / 64, 256, 0, stream>>>(content0, nid0, emb, wpb0, bp0, h0);
  embed_kernel<<<N1 / 64, 256, 0, stream>>>(content1, nid1, emb, wpb1, bp1, h1e);
  embed_kernel<<<N2 / 64, 256, 0, stream>>>(content2, nid2, emb, wpb2, bp2, h2e);

  long long t0 = (long long)E0 * FDIM;
  scatter_kernel<<<(int)((t0 + 255) / 256), 256, 0, stream>>>(h0, b0s, b0d, agg0, cnt0, E0);
  conv_kernel<<<N1 / 64, 256, 0, stream>>>(h1e, agg0, cnt0, wcb0, bc0, h1);

  long long t1 = (long long)E1 * FDIM;
  scatter_kernel<<<(int)((t1 + 255) / 256), 256, 0, stream>>>(h1, b1s, b1d, agg1, cnt1, E1);
  conv_kernel<<<N2 / 64, 256, 0, stream>>>(h2e, agg1, cnt1, wcb1, bc1, (float*)d_out);
}
